// SimilarIntent_2388001816921
// MI455X (gfx1250) — matrix operand order repaired and checked by reading, not compiled, not run on hardware
//
#include <hip/hip_runtime.h>
#include <hip/hip_bf16.h>

// ---------------------------------------------------------------------------
// SimilarIntent: h [N,D] fp32 -> row-normalize -> sim = hn@hn^T -> top-10
//                -> softmax(12*sim_topk) -> weighted gather of h rows.
// N=16384, D=128. Fused, sim never materialized, barrier-free main loop.
// Software-pipelined: two live B register sets so next tile's 8x
// global_load_b128 overlap the current tile's 4x v_wmma_f32_16x16x32_f16
// and the top-10 scan (partial loadcnt waits instead of full stalls).
// ---------------------------------------------------------------------------

#define NROWS 16384
#define DIM   128
#define TOPK  10
#define THETA 12.0f
#define MTILE 128
#define WAVES 8

typedef __attribute__((ext_vector_type(16))) _Float16 v16h;
typedef __attribute__((ext_vector_type(8)))  _Float16 v8h;
typedef __attribute__((ext_vector_type(4)))  _Float16 v4h;
typedef __attribute__((ext_vector_type(8)))  float    v8f;

static __device__ __forceinline__ v16h join8(v8h lo, v8h hi) {
  return __builtin_shufflevector(lo, hi, 0, 1, 2, 3, 4, 5, 6, 7,
                                         8, 9, 10, 11, 12, 13, 14, 15);
}

static __device__ __forceinline__ void wave_lds_fence() {
  __builtin_amdgcn_wave_barrier();                 // compiler ordering point
  asm volatile("s_wait_dscnt 0x0" ::: "memory");   // CDNA5 split DS counter
}

struct BReg { v16h b[4]; };

// ---------------------------------------------------------------------------
// Kernel 1: row-normalize h (fp32) into hn (f16). One wave32 per row.
// ---------------------------------------------------------------------------
__global__ __launch_bounds__(256) void si_rownorm_kernel(
    const float* __restrict__ h, _Float16* __restrict__ hn) {
  const int row  = (blockIdx.x << 3) + (threadIdx.x >> 5);
  const int lane = threadIdx.x & 31;

  const float4 v = *(const float4*)(h + (size_t)row * DIM + lane * 4);
  float ss = v.x * v.x + v.y * v.y + v.z * v.z + v.w * v.w;
#pragma unroll
  for (int m = 16; m > 0; m >>= 1) ss += __shfl_xor(ss, m, 32);

  const float inv = 1.0f / fmaxf(sqrtf(ss), 1e-8f);
  v4h o;
  o[0] = (_Float16)(v.x * inv);
  o[1] = (_Float16)(v.y * inv);
  o[2] = (_Float16)(v.z * inv);
  o[3] = (_Float16)(v.w * inv);
  *(v4h*)(hn + (size_t)row * DIM + lane * 4) = o;
}

// ---------------------------------------------------------------------------
// Kernel 2: fused WMMA similarity + streaming top-10 + softmax + aggregation.
// ---------------------------------------------------------------------------
__global__ __launch_bounds__(256) void si_simtopk_kernel(
    const float* __restrict__ h, const _Float16* __restrict__ hn,
    float* __restrict__ out) {
  __shared__ float s_tile[WAVES][16][17];           // per-wave C tile, padded
  __shared__ float s_mv[WAVES][16][2][TOPK];        // per-row partial lists
  __shared__ int   s_mi[WAVES][16][2][TOPK];
  __shared__ float s_beta[WAVES][16][TOPK];
  __shared__ int   s_bidx[WAVES][16][TOPK];

  const int tid     = threadIdx.x;
  const int lane    = tid & 31;
  const int wid     = tid >> 5;
  const int rowBase = blockIdx.x * MTILE + wid * 16;

  // ---- A fragment: this wave's 16 rows x 128 K, register-resident ----
  // A 16x32 k order used here (contiguous): lanes 0-15 (row=lane)   hold K {0-15},
  //                            lanes 16-31 (row=lane-16) hold K {16-31}.
  const int amrow = lane & 15;
  const int khalf = (lane < 16) ? 0 : 16;
  const _Float16* arow = hn + (size_t)(rowBase + amrow) * DIM;
  v16h afrag[4];
#pragma unroll
  for (int kk = 0; kk < 4; ++kk) {
    v8h lo = *(const v8h*)(arow + kk * 32 + khalf);
    v8h hi = *(const v8h*)(arow + kk * 32 + 8 + khalf);
    afrag[kk] = join8(lo, hi);
  }

  // ---- private top-10 (descending), half a row per lane ----
  float tv[TOPK];
  int   ti[TOPK];
#pragma unroll
  for (int k = 0; k < TOPK; ++k) { tv[k] = -3.0e38f; ti[k] = -1; }

  // B layout (32x16 f16): lanes 0-15 hold K=0..15 of col n=lane,
  //                       lanes 16-31 hold K=16..31 of col n=lane-16.
  const int bn      = lane & 15;
  const int bkoff   = (lane < 16) ? 0 : 16;
  const int rb      = (lane < 16) ? 0 : 8;   // C rows this lane holds
  const int scanRow = lane >> 1;             // 0..15: row this lane scans
  const int scanC0  = (lane & 1) * 8;        // half-row column offset

  const _Float16* bbase = hn + (size_t)bn * DIM + bkoff;

  auto loadB = [&](int colBase) -> BReg {
    const _Float16* brow = bbase + (size_t)colBase * DIM;
    v8h q0 = *(const v8h*)(brow);
    v8h q1 = *(const v8h*)(brow + 8);
    v8h q2 = *(const v8h*)(brow + 32);
    v8h q3 = *(const v8h*)(brow + 40);
    v8h q4 = *(const v8h*)(brow + 64);
    v8h q5 = *(const v8h*)(brow + 72);
    v8h q6 = *(const v8h*)(brow + 96);
    v8h q7 = *(const v8h*)(brow + 104);
    BReg r;
    r.b[0] = join8(q0, q1);
    r.b[1] = join8(q2, q3);
    r.b[2] = join8(q4, q5);
    r.b[3] = join8(q6, q7);
    return r;
  };

  auto processTile = [&](int colBase, const BReg& bf) {
    v8f c = {0.f, 0.f, 0.f, 0.f, 0.f, 0.f, 0.f, 0.f};
    c = __builtin_amdgcn_wmma_f32_16x16x32_f16(false, afrag[0], false, bf.b[0],
                                               (short)0, c, false, false);
    c = __builtin_amdgcn_wmma_f32_16x16x32_f16(false, afrag[1], false, bf.b[1],
                                               (short)0, c, false, false);
    c = __builtin_amdgcn_wmma_f32_16x16x32_f16(false, afrag[2], false, bf.b[2],
                                               (short)0, c, false, false);
    c = __builtin_amdgcn_wmma_f32_16x16x32_f16(false, afrag[3], false, bf.b[3],
                                               (short)0, c, false, false);

    // wave-private C tile -> LDS (row-major view for the scan)
#pragma unroll
    for (int r = 0; r < 8; ++r) s_tile[wid][rb + r][bn] = c[r];
    wave_lds_fence();

    // streaming top-10: each lane scans 8 candidates of its half-row.
    // Outer reject test stays branchy (cheap skip); insertion is branchless
    // cndmask network to avoid divergent code explosion.
#pragma unroll
    for (int q = 0; q < 8; ++q) {
      const float v = s_tile[wid][scanRow][scanC0 + q];
      if (v > tv[TOPK - 1]) {
        float cv = v;
        int   ci = colBase + scanC0 + q;
#pragma unroll
        for (int p = 0; p < TOPK; ++p) {
          const bool  sw = cv > tv[p];
          const float ov = tv[p];
          const int   oi = ti[p];
          tv[p] = sw ? cv : ov;
          ti[p] = sw ? ci : oi;
          cv    = sw ? ov : cv;
          ci    = sw ? oi : ci;
        }
      }
    }
    wave_lds_fence();   // scan loads done before next tile's stores
  };

  // ---- software-pipelined sweep over all 16384 columns ----
  BReg bcur = loadB(0);
  for (int colBase = 0; colBase < NROWS; colBase += 32) {
    if (colBase + 512 < NROWS)               // warm L2->WGP for later tiles
      __builtin_prefetch(bbase + (size_t)(colBase + 512) * DIM, 0, 1);

    BReg bnext = loadB(colBase + 16);        // in flight during tile 0
    processTile(colBase, bcur);
    bcur = loadB((colBase + 32) & (NROWS - 1));  // wrap: branch-free tail
    processTile(colBase + 16, bnext);
  }

  // ---- merge the two half-row lists per row (wave-local) ----
  const int hh = lane & 1;
#pragma unroll
  for (int k = 0; k < TOPK; ++k) {
    s_mv[wid][scanRow][hh][k] = tv[k];
    s_mi[wid][scanRow][hh][k] = ti[k];
  }
  wave_lds_fence();

  if (lane < 16) {
    float fv[TOPK];
    int   fi[TOPK];
#pragma unroll
    for (int k = 0; k < TOPK; ++k) { fv[k] = -3.0e38f; fi[k] = -1; }
    const float* pv = &s_mv[wid][lane][0][0];
    const int*   pi = &s_mi[wid][lane][0][0];
#pragma unroll
    for (int e = 0; e < 2 * TOPK; ++e) {
      float cv = pv[e];
      int   ci = pi[e];
#pragma unroll
      for (int p = 0; p < TOPK; ++p) {
        const bool  sw = cv > fv[p];
        const float ov = fv[p];
        const int   oi = fi[p];
        fv[p] = sw ? cv : ov;
        fi[p] = sw ? ci : oi;
        cv    = sw ? ov : cv;
        ci    = sw ? oi : ci;
      }
    }
    // softmax(THETA * topk), max-subtracted, fp32
    const float mx = fv[0];
    float w[TOPK], s = 0.f;
#pragma unroll
    for (int k = 0; k < TOPK; ++k) { w[k] = __expf(THETA * (fv[k] - mx)); s += w[k]; }
    const float inv = 1.0f / s;
#pragma unroll
    for (int k = 0; k < TOPK; ++k) {
      s_beta[wid][lane][k] = w[k] * inv;
      s_bidx[wid][lane][k] = fi[k];
    }
  }
  wave_lds_fence();

  // ---- weighted aggregation of original fp32 h rows ----
  // 2 lanes per row, 64 dims each, processed in 8-dim chunks.
  float bw[TOPK];
  int   bi[TOPK];
#pragma unroll
  for (int k = 0; k < TOPK; ++k) {
    bw[k] = s_beta[wid][scanRow][k];
    bi[k] = s_bidx[wid][scanRow][k];
  }
  const int dbase = (lane & 1) * 64;
  float* orow = out + (size_t)(rowBase + scanRow) * DIM + dbase;
#pragma unroll
  for (int d0 = 0; d0 < 64; d0 += 8) {
    float4 a0 = {0.f, 0.f, 0.f, 0.f};
    float4 a1 = {0.f, 0.f, 0.f, 0.f};
#pragma unroll
    for (int k = 0; k < TOPK; ++k) {
      const float* hr = h + (size_t)bi[k] * DIM + dbase + d0;
      const float4 x0 = *(const float4*)hr;
      const float4 x1 = *(const float4*)(hr + 4);
      a0.x += bw[k] * x0.x; a0.y += bw[k] * x0.y;
      a0.z += bw[k] * x0.z; a0.w += bw[k] * x0.w;
      a1.x += bw[k] * x1.x; a1.y += bw[k] * x1.y;
      a1.z += bw[k] * x1.z; a1.w += bw[k] * x1.w;
    }
    *(float4*)(orow + d0)     = a0;
    *(float4*)(orow + d0 + 4) = a1;
  }
}

// ---------------------------------------------------------------------------
extern "C" void kernel_launch(void* const* d_in, const int* in_sizes, int n_in,
                              void* d_out, int out_size, void* d_ws,
                              size_t ws_size, hipStream_t stream) {
  (void)in_sizes; (void)n_in; (void)out_size; (void)ws_size;
  const float* h   = (const float*)d_in[0];
  float*       out = (float*)d_out;
  _Float16*    hn  = (_Float16*)d_ws;  // N*D f16 = 4 MB scratch

  si_rownorm_kernel<<<NROWS / 8, 256, 0, stream>>>(h, hn);
  si_simtopk_kernel<<<NROWS / MTILE, 256, 0, stream>>>(h, hn, out);
}
